// AFT_FULL_43069932044795
// MI455X (gfx1250) — matrix operand order repaired and checked by reading, not compiled, not run on hardware
//
#include <hip/hip_runtime.h>
#include <math.h>

typedef _Float16 half8  __attribute__((ext_vector_type(8)));
typedef _Float16 half16 __attribute__((ext_vector_type(16)));
typedef float    floatx8 __attribute__((ext_vector_type(8)));

static constexpr int Cdim = 128;
static constexpr int Ndim = 4096;

__device__ inline half16 cat8(half8 lo, half8 hi) {
  return __builtin_shufflevector(lo, hi, 0,1,2,3,4,5,6,7,8,9,10,11,12,13,14,15);
}

// Workgroup-relative LDS byte address = low 32 bits of the generic pointer
__device__ inline unsigned lds_addr32(const void* p) { return (unsigned)(size_t)p; }

// CDNA5 async memory->LDS copy, 16B per lane, tracked by ASYNCcnt.
#define ASYNC_LD_B128(ldsdst, gaddr) \
  asm volatile("global_load_async_to_lds_b128 %0, %1, off" \
               :: "v"(ldsdst), "v"(gaddr) : "memory")
#define WAIT_ASYNC0() asm volatile("s_wait_asynccnt 0x0" ::: "memory")

// ---------------------------------------------------------------------------
// Pack [Wq;Wk;Wv] -> W16 (384 x 128) f16
// ---------------------------------------------------------------------------
__global__ void wpack_kernel(const float* __restrict__ Wq, const float* __restrict__ Wk,
                             const float* __restrict__ Wv, _Float16* __restrict__ W16) {
  int idx = blockIdx.x * blockDim.x + threadIdx.x;     // 0 .. 49151
  if (idx >= 3 * Cdim * Cdim) return;
  float v;
  if (idx < Cdim * Cdim)            v = Wq[idx];
  else if (idx < 2 * Cdim * Cdim)   v = Wk[idx - Cdim * Cdim];
  else                              v = Wv[idx - 2 * Cdim * Cdim];
  W16[idx] = (_Float16)v;
}

// ---------------------------------------------------------------------------
// eB16[t][s] = exp(B[t][s] - max_s B[t][s])  (row-max cancels in num/den)
// ---------------------------------------------------------------------------
__global__ __launch_bounds__(256)
void rowmax_exp_kernel(const float* __restrict__ Bm, _Float16* __restrict__ eB) {
  const int t = blockIdx.x, tid = threadIdx.x;
  const float* row = Bm + (size_t)t * Ndim;
  float v[16]; float m = -INFINITY;
  #pragma unroll
  for (int i = 0; i < 16; ++i) { v[i] = row[tid + i * 256]; m = fmaxf(m, v[i]); }
  __shared__ float red[256];
  red[tid] = m; __syncthreads();
  for (int s = 128; s > 0; s >>= 1) { if (tid < s) red[tid] = fmaxf(red[tid], red[tid + s]); __syncthreads(); }
  m = red[0];
  _Float16* orow = eB + (size_t)t * Ndim;
  #pragma unroll
  for (int i = 0; i < 16; ++i) orow[tid + i * 256] = (_Float16)expf(v[i] - m);
}

// ---------------------------------------------------------------------------
// qkv GEMM:  [Wq;Wk;Wv](384x128 f16)  @  x[b](128x4096 -> f16 in LDS)
// Output lands transposed & coalesced: qT/kT/vT [b][c][n] f32 (+bias)
// ---------------------------------------------------------------------------
__global__ __launch_bounds__(256)
void qkv_kernel(const float* __restrict__ x, const _Float16* __restrict__ W16,
                const float* __restrict__ bq, const float* __restrict__ bk,
                const float* __restrict__ bv,
                float* __restrict__ qT, float* __restrict__ kT, float* __restrict__ vT) {
  const int b   = blockIdx.y;
  const int n0  = blockIdx.x * 64;
  const int tid = threadIdx.x;
  const int lane = tid & 31;
  const int wv   = tid >> 5;        // 0..7 (wave32)
  const int lh   = lane & 15;
  const int kh   = lane >> 4;

  __shared__ _Float16 Xs[64 * 136]; // [n][c_in], stride 136 halves (bank-friendly)

  { // stage x tile (coalesced f32 reads, f16 LDS)
    const int n  = tid & 63;
    const int cb = tid >> 6;        // 0..3
    const float* xb = x + (size_t)b * Cdim * Ndim + n0 + n;
    for (int c = cb; c < Cdim; c += 4)
      Xs[n * 136 + c] = (_Float16)xb[(size_t)c * Ndim];
  }
  __syncthreads();

  floatx8 acc[3][4];
  #pragma unroll
  for (int mi = 0; mi < 3; ++mi)
    #pragma unroll
    for (int ni = 0; ni < 4; ++ni)
      #pragma unroll
      for (int e = 0; e < 8; ++e) acc[mi][ni][e] = 0.0f;

  #pragma unroll
  for (int ks = 0; ks < 4; ++ks) {                 // K = 128 in steps of 32
    half16 af[3];
    #pragma unroll
    for (int mi = 0; mi < 3; ++mi) {               // A frag: lane=M row, split-K halves
      const int row = wv * 48 + mi * 16 + lh;
      const _Float16* wp = W16 + row * Cdim + ks * 32 + kh * 8;
      af[mi] = cat8(*(const half8*)wp, *(const half8*)(wp + 16));
    }
    #pragma unroll
    for (int ni = 0; ni < 4; ++ni) {               // B frag: lane=N col, split-K halves
      const int rowb = (ni * 16 + lh) * 136 + ks * 32 + kh * 8;
      half16 bf = cat8(*(const half8*)&Xs[rowb], *(const half8*)&Xs[rowb + 16]);
      #pragma unroll
      for (int mi = 0; mi < 3; ++mi)
        acc[mi][ni] = __builtin_amdgcn_wmma_f32_16x16x32_f16(
            false, af[mi], false, bf, (short)0, acc[mi][ni], false, false);
    }
  }

  // epilogue: D lane = n column, VGPR m = c_out row (m / m+8)
  #pragma unroll
  for (int mi = 0; mi < 3; ++mi)
    #pragma unroll
    for (int ni = 0; ni < 4; ++ni) {
      const int n = n0 + ni * 16 + lh;
      #pragma unroll
      for (int m = 0; m < 8; ++m) {
        const int r = wv * 48 + mi * 16 + m + kh * 8;
        float* dst; const float* bias; int rr;
        if (r < Cdim)          { dst = qT; bias = bq; rr = r; }
        else if (r < 2 * Cdim) { dst = kT; bias = bk; rr = r - Cdim; }
        else                   { dst = vT; bias = bv; rr = r - 2 * Cdim; }
        dst[((size_t)b * Cdim + rr) * Ndim + n] = acc[mi][ni][m] + bias[rr];
      }
    }
}

// ---------------------------------------------------------------------------
// kmax[b][c] = max_s kT[b][c][s]   (per-(b,c) scaling cancels in num/den)
// ---------------------------------------------------------------------------
__global__ __launch_bounds__(256)
void kmax_kernel(const float* __restrict__ kT, float* __restrict__ kmax) {
  const int c = blockIdx.x, b = blockIdx.y, tid = threadIdx.x;
  const float* row = kT + ((size_t)b * Cdim + c) * Ndim;
  float m = -INFINITY;
  #pragma unroll
  for (int i = 0; i < 16; ++i) m = fmaxf(m, row[tid + i * 256]);
  __shared__ float red[256];
  red[tid] = m; __syncthreads();
  for (int s = 128; s > 0; s >>= 1) { if (tid < s) red[tid] = fmaxf(red[tid], red[tid + s]); __syncthreads(); }
  if (tid == 0) kmax[b * Cdim + c] = red[0];
}

// ---------------------------------------------------------------------------
// PT[b][2c][s] = f16(exp(k-kmax)*v)   (numerator rows, even)
// PT[b][2c+1][s] = f16(exp(k-kmax))   (denominator rows, odd)
// ---------------------------------------------------------------------------
__global__ __launch_bounds__(256)
void ptbuild_kernel(const float* __restrict__ kT, const float* __restrict__ vT,
                    const float* __restrict__ kmax, _Float16* __restrict__ PT) {
  const int c = blockIdx.x, b = blockIdx.y, tid = threadIdx.x;
  const float* krow = kT + ((size_t)b * Cdim + c) * Ndim;
  const float* vrow = vT + ((size_t)b * Cdim + c) * Ndim;
  const float  km   = kmax[b * Cdim + c];
  _Float16* numrow = PT + ((size_t)b * 2 * Cdim + 2 * c)     * Ndim;
  _Float16* denrow = PT + ((size_t)b * 2 * Cdim + 2 * c + 1) * Ndim;
  #pragma unroll
  for (int i = 0; i < 16; ++i) {
    const int s = tid + i * 256;
    const float ek = expf(krow[s] - km);
    denrow[s] = (_Float16)ek;
    numrow[s] = (_Float16)(ek * vrow[s]);
  }
}

// ---------------------------------------------------------------------------
// Main GEMM per batch:  D[c2][t] = sum_s PT[c2][s] * eB[t][s]   (c2 = 256 rows)
// Block: 256(c2) x 128(t); K in 32-chunks; double-buffered LDS filled by
// CDNA5 async memory->LDS copies (global_load_async_to_lds_b128 / ASYNCcnt).
// Epilogue fuses sigmoid(q) * num/den, coalesced stores to out[b][c][t].
// ---------------------------------------------------------------------------
__global__ __launch_bounds__(512)
void aft_attn_kernel(const _Float16* __restrict__ PT, const _Float16* __restrict__ eB,
                     const float* __restrict__ qT, float* __restrict__ out) {
  const int b   = blockIdx.y;
  const int t0  = blockIdx.x * 128;
  const int tid = threadIdx.x;
  const int lane = tid & 31;
  const int wv   = tid >> 5;   // 0..15
  const int mg   = wv & 3;     // c2-group: 64 rows
  const int ng   = wv >> 2;    // t-group : 32 cols
  const int lh   = lane & 15;
  const int kh   = lane >> 4;

  __shared__ _Float16 As[2][256 * 40];  // PT tile  [c2][32s], stride 40 halves
  __shared__ _Float16 Bs[2][128 * 40];  // eB tile  [t ][32s], stride 40 halves

  // --- async staging assignment: 3x 16B chunks per thread per K-step -------
  // A tile: 256 rows x 4 chunks = 1024 chunks -> chunks tid and tid+512
  // B tile: 128 rows x 4 chunks =  512 chunks -> chunk tid
  const int qa0 = tid,        ra0 = qa0 >> 2, pa0 = qa0 & 3;
  const int qa1 = tid + 512,  ra1 = qa1 >> 2, pa1 = qa1 & 3;
  const int rb  = tid >> 2,   pb  = tid & 3;

  unsigned long long gaA0 = (unsigned long long)(const char*)(PT + ((size_t)b * 256 + ra0) * Ndim + pa0 * 8);
  unsigned long long gaA1 = (unsigned long long)(const char*)(PT + ((size_t)b * 256 + ra1) * Ndim + pa1 * 8);
  unsigned long long gaB  = (unsigned long long)(const char*)(eB + ((size_t)(t0 + rb)) * Ndim + pb * 8);

  unsigned ldsA0[2], ldsA1[2], ldsB[2];
  #pragma unroll
  for (int u = 0; u < 2; ++u) {
    ldsA0[u] = lds_addr32(&As[u][ra0 * 40 + pa0 * 8]);
    ldsA1[u] = lds_addr32(&As[u][ra1 * 40 + pa1 * 8]);
    ldsB[u]  = lds_addr32(&Bs[u][rb  * 40 + pb  * 8]);
  }

  // prologue: fill buffer 0 (K-chunk 0)
  ASYNC_LD_B128(ldsA0[0], gaA0);
  ASYNC_LD_B128(ldsA1[0], gaA1);
  ASYNC_LD_B128(ldsB[0],  gaB);
  gaA0 += 64; gaA1 += 64; gaB += 64;
  WAIT_ASYNC0();
  __syncthreads();

  floatx8 acc[4][2];
  #pragma unroll
  for (int mi = 0; mi < 4; ++mi)
    #pragma unroll
    for (int ni = 0; ni < 2; ++ni)
      #pragma unroll
      for (int e = 0; e < 8; ++e) acc[mi][ni][e] = 0.0f;

  const int KSTEPS = Ndim / 32;
  #pragma unroll 2
  for (int kk = 0; kk < KSTEPS; ++kk) {
    const int buf = kk & 1;
    if (kk + 1 < KSTEPS) {                 // kick off async fill of other buffer
      const int nb = buf ^ 1;
      ASYNC_LD_B128(ldsA0[nb], gaA0);
      ASYNC_LD_B128(ldsA1[nb], gaA1);
      ASYNC_LD_B128(ldsB[nb],  gaB);
      gaA0 += 64; gaA1 += 64; gaB += 64;
    }
    half16 bf[2];
    #pragma unroll
    for (int ni = 0; ni < 2; ++ni) {       // B frag: lane = t col, split-K halves
      const int rowb = (ng * 32 + ni * 16 + lh) * 40 + kh * 8;
      bf[ni] = cat8(*(const half8*)&Bs[buf][rowb], *(const half8*)&Bs[buf][rowb + 16]);
    }
    #pragma unroll
    for (int mi = 0; mi < 4; ++mi) {       // A frag: lane = c2 row, split-K halves
      const int rowa = (mg * 64 + mi * 16 + lh) * 40 + kh * 8;
      half16 af = cat8(*(const half8*)&As[buf][rowa], *(const half8*)&As[buf][rowa + 16]);
      #pragma unroll
      for (int ni = 0; ni < 2; ++ni)
        acc[mi][ni] = __builtin_amdgcn_wmma_f32_16x16x32_f16(
            false, af, false, bf[ni], (short)0, acc[mi][ni], false, false);
    }
    WAIT_ASYNC0();                         // own async copies landed in LDS
    __syncthreads();                       // everyone's copies visible
  }

  // Epilogue: even VGPR = numerator row, odd VGPR = denominator row (same lane).
  #pragma unroll
  for (int mi = 0; mi < 4; ++mi)
    #pragma unroll
    for (int ni = 0; ni < 2; ++ni) {
      const int t = t0 + ng * 32 + ni * 16 + lh;
      #pragma unroll
      for (int i = 0; i < 4; ++i) {
        const float num = acc[mi][ni][2 * i];
        const float den = acc[mi][ni][2 * i + 1];
        const int c = mg * 32 + mi * 8 + kh * 4 + i;
        const size_t idx = ((size_t)b * Cdim + c) * Ndim + t;
        const float q = qT[idx];
        const float sig = 1.0f / (1.0f + expf(-q));
        out[idx] = sig * (num / den);
      }
    }
}

// ---------------------------------------------------------------------------
extern "C" void kernel_launch(void* const* d_in, const int* in_sizes, int n_in,
                              void* d_out, int out_size, void* d_ws, size_t ws_size,
                              hipStream_t stream) {
  (void)in_sizes; (void)n_in; (void)out_size; (void)ws_size;
  const float* x  = (const float*)d_in[0];
  const float* Wq = (const float*)d_in[1];
  const float* bq = (const float*)d_in[2];
  const float* Wk = (const float*)d_in[3];
  const float* bk = (const float*)d_in[4];
  const float* Wv = (const float*)d_in[5];
  const float* bv = (const float*)d_in[6];
  const float* pb = (const float*)d_in[7];
  float* out = (float*)d_out;

  char* ws = (char*)d_ws;
  size_t off = 0;
  _Float16* W16 = (_Float16*)(ws + off); off += (size_t)3 * Cdim * Cdim * sizeof(_Float16);
  off = (off + 255) & ~(size_t)255;
  float* qT   = (float*)(ws + off);   off += (size_t)8 * Cdim * Ndim * sizeof(float);
  float* kT   = (float*)(ws + off);   off += (size_t)8 * Cdim * Ndim * sizeof(float);
  float* vT   = (float*)(ws + off);   off += (size_t)8 * Cdim * Ndim * sizeof(float);
  float* kmax = (float*)(ws + off);   off += (size_t)8 * Cdim * sizeof(float);
  off = (off + 255) & ~(size_t)255;
  _Float16* eB16 = (_Float16*)(ws + off); off += (size_t)Ndim * Ndim * sizeof(_Float16);
  _Float16* PT   = (_Float16*)(ws + off); off += (size_t)8 * 2 * Cdim * Ndim * sizeof(_Float16);

  wpack_kernel<<<(3 * Cdim * Cdim + 255) / 256, 256, 0, stream>>>(Wq, Wk, Wv, W16);
  rowmax_exp_kernel<<<Ndim, 256, 0, stream>>>(pb, eB16);
  qkv_kernel<<<dim3(Ndim / 64, 8), 256, 0, stream>>>(x, W16, bq, bk, bv, qT, kT, vT);
  kmax_kernel<<<dim3(Cdim, 8), 256, 0, stream>>>(kT, kmax);
  ptbuild_kernel<<<dim3(Cdim, 8), 256, 0, stream>>>(kT, vT, kmax, PT);
  aft_attn_kernel<<<dim3(Ndim / 128, 8), 512, 0, stream>>>(PT, eB16, qT, out);
}
